// SwinTransformer_36507222016004
// MI455X (gfx1250) — matrix operand order repaired and checked by reading, not compiled, not run on hardware
//
#include <hip/hip_runtime.h>

// ---------------------------------------------------------------------------
// Swin block x2 for gfx1250 (MI455X): bf16 WMMA everywhere, fp32 accumulate.
// GEMM tiles staged through LDS with a double-buffered ASYNCcnt pipeline
// (GLOBAL_LOAD_ASYNC_TO_LDS_B128); falls back to synchronous staging if the
// async builtins are unavailable.
// ---------------------------------------------------------------------------

typedef __bf16 bf16_t;
typedef __attribute__((ext_vector_type(16))) __bf16 v16bf;
typedef __attribute__((ext_vector_type(8)))  float  v8f;
typedef __attribute__((ext_vector_type(4)))  int    v4i;

#define B_     32
#define H_     56
#define C_     192
#define NH_    6
#define WS_    7
#define N_     49
#define HD_    32
#define BW_    (B_ * 64)        // 2048 windows total (8x8 per image)
#define MTOT   (BW_ * N_)       // 100352 token rows
#define SCALE_ 0.1767766952966369f

#if defined(__gfx1250__) && \
    __has_builtin(__builtin_amdgcn_global_load_async_to_lds_b128) && \
    __has_builtin(__builtin_amdgcn_s_wait_asynccnt)
#define HAVE_ASYNC_LDS 1
#else
#define HAVE_ASYNC_LDS 0
#endif

#define AS1 __attribute__((address_space(1)))
#define AS3 __attribute__((address_space(3)))

union Frag { v16bf v; uint4 q[2]; };

__device__ __forceinline__ bf16_t f2bf(float f) { return (bf16_t)f; }

#if HAVE_ASYNC_LDS
// one 16-byte async global->LDS transfer per lane (ASYNCcnt-tracked)
__device__ __forceinline__ void async_copy16(const bf16_t* g, bf16_t* l) {
    __builtin_amdgcn_global_load_async_to_lds_b128(
        (AS1 v4i*)g, (AS3 v4i*)l, 0, 0);
}
#endif

// region id for the shifted-window attention mask (shift = WS/2 = 3)
__device__ __forceinline__ int regid(int h) {
    return (h < H_ - WS_) ? 0 : ((h < H_ - 3) ? 1 : 2);
}

// ---------------------------------------------------------------------------
// fp32 -> bf16 weight conversion
// ---------------------------------------------------------------------------
__global__ void cvt_kernel(const float* __restrict__ s, bf16_t* __restrict__ d, int n) {
    int i = blockIdx.x * 256 + threadIdx.x;
    if (i < n) d[i] = f2bf(s[i]);
}

// ---------------------------------------------------------------------------
// LayerNorm (one wave32 per token). WINDOW=true: also shift + window-partition
// into (win, n) row order. Output bf16 [MTOT][C].
// ---------------------------------------------------------------------------
template<bool WINDOW>
__global__ __launch_bounds__(256) void ln_kernel(
    const float* __restrict__ x, const float* __restrict__ gw,
    const float* __restrict__ gb, bf16_t* __restrict__ out, int shift)
{
    int wave = blockIdx.x * 8 + (threadIdx.x >> 5);
    int lane = threadIdx.x & 31;
    if (wave >= MTOT) return;

    int src;
    if (WINDOW) {
        int win  = wave / N_, nidx = wave % N_;
        int bimg = win >> 6,  wrem = win & 63;
        int wy = wrem >> 3, wx = wrem & 7;
        int py = nidx / WS_, px = nidx % WS_;
        int sh = (wy * WS_ + py + shift) % H_;   // roll(-shift): out[i]=in[i+shift]
        int sw = (wx * WS_ + px + shift) % H_;
        src = bimg * (H_ * H_) + sh * H_ + sw;
    } else {
        src = wave;
    }

    const float* xp = x + (size_t)src * C_;
    float v[6];
    float s = 0.f;
#pragma unroll
    for (int i = 0; i < 6; i++) { v[i] = xp[lane + 32 * i]; s += v[i]; }
#pragma unroll
    for (int o = 16; o >= 1; o >>= 1) s += __shfl_xor(s, o, 32);
    float mu = s * (1.f / C_);
    float vs = 0.f;
#pragma unroll
    for (int i = 0; i < 6; i++) { float d = v[i] - mu; vs += d * d; }
#pragma unroll
    for (int o = 16; o >= 1; o >>= 1) vs += __shfl_xor(vs, o, 32);
    float rstd = rsqrtf(vs * (1.f / C_) + 1e-5f);

    bf16_t* op = out + (size_t)wave * C_;
#pragma unroll
    for (int i = 0; i < 6; i++) {
        int c = lane + 32 * i;
        op[c] = f2bf((v[i] - mu) * rstd * gw[c] + gb[c]);
    }
}

// ---------------------------------------------------------------------------
// Generic bf16 GEMM:  out[M][Nout] = A[M][K] * W[Nout][K]^T + bias
// Block tile 128x64, 8 waves, each wave 32x32 (2x2 WMMA 16x16), BK=32.
// MODE 0: store bf16    MODE 1: GELU -> bf16
// MODE 2: proj epilogue (window-reverse + quirky roll(+WS) + residual into x)
// MODE 3: MLP2 epilogue (token-order residual, dst may be d_out)
// ---------------------------------------------------------------------------
#define BM  128
#define BN  64
#define BK  32
#define LDA 40   // LDS row stride in bf16 elements (80B, 16B aligned)

#if HAVE_ASYNC_LDS
#define NBUF 2
#else
#define NBUF 1
#endif

template<int MODE>
__global__ __launch_bounds__(256) void gemm_kernel(
    const bf16_t* __restrict__ A, const bf16_t* __restrict__ W,
    const float* __restrict__ bias, int M, int K, int Nout,
    const float* __restrict__ xsrc, float* __restrict__ xdst,
    bf16_t* __restrict__ out_bf, int shift)
{
    __shared__ bf16_t As[NBUF][BM * LDA];
    __shared__ bf16_t Bs[NBUF][BN * LDA];

    int tid  = threadIdx.x;
    int lane = tid & 31, wv = tid >> 5;
    int m0 = blockIdx.y * BM, n0 = blockIdx.x * BN;
    int wm = (wv & 3) * 32, wn = (wv >> 2) * 32;
    int half = lane >> 4, l16 = lane & 15;

    v8f acc[2][2];
    v8f zero = {0.f, 0.f, 0.f, 0.f, 0.f, 0.f, 0.f, 0.f};
    acc[0][0] = zero; acc[0][1] = zero; acc[1][0] = zero; acc[1][1] = zero;

    int arow = tid >> 1, akc = (tid & 1) * 16;   // A: 2 threads/row, 16 el each
    int brow = tid >> 2, bkc = (tid & 3) * 8;    // B: 4 threads/row, 8 el each

    // WMMA compute from LDS buffer b (fragments per the 16x16x32 bf16 layout)
    auto compute_tile = [&](int b) {
        Frag a[2], bb[2];
#pragma unroll
        for (int tm = 0; tm < 2; tm++) {
            const bf16_t* p = &As[b][(wm + tm * 16 + l16) * LDA + half * 8];
            a[tm].q[0] = *(const uint4*)p;        // K chunk [half*8, +8)
            a[tm].q[1] = *(const uint4*)(p + 16); // K chunk [16+half*8, +8)
        }
#pragma unroll
        for (int tn = 0; tn < 2; tn++) {
            const bf16_t* p = &Bs[b][(wn + tn * 16 + l16) * LDA + half * 8];
            bb[tn].q[0] = *(const uint4*)p;       // K chunk [half*8, +8)
            bb[tn].q[1] = *(const uint4*)(p + 16);
        }
#pragma unroll
        for (int tm = 0; tm < 2; tm++)
#pragma unroll
            for (int tn = 0; tn < 2; tn++)
                acc[tm][tn] = __builtin_amdgcn_wmma_f32_16x16x32_bf16(
                    false, a[tm].v, false, bb[tn].v, (short)0, acc[tm][tn],
                    false, false);
    };

#if HAVE_ASYNC_LDS
    // -------- double-buffered async global->LDS pipeline (ASYNCcnt) --------
    auto issue_tile = [&](int k0, int b) {
        const bf16_t* ag = A + (size_t)(m0 + arow) * K + k0 + akc;
        const bf16_t* bg = W + (size_t)(n0 + brow) * K + k0 + bkc;
        async_copy16(ag,     &As[b][arow * LDA + akc]);
        async_copy16(ag + 8, &As[b][arow * LDA + akc + 8]);
        async_copy16(bg,     &Bs[b][brow * LDA + bkc]);
    };

    int buf = 0;
    issue_tile(0, buf);
    __builtin_amdgcn_s_wait_asynccnt(0);
    __syncthreads();
    for (int k0 = 0; k0 < K; k0 += BK) {
        if (k0 + BK < K) issue_tile(k0 + BK, buf ^ 1);  // prefetch next tile
        compute_tile(buf);
        __builtin_amdgcn_s_wait_asynccnt(0);            // next tile landed
        __syncthreads();
        buf ^= 1;
    }
#else
    // -------- synchronous staging fallback --------
    for (int k0 = 0; k0 < K; k0 += BK) {
        const bf16_t* ag = A + (size_t)(m0 + arow) * K + k0 + akc;
        const bf16_t* bg = W + (size_t)(n0 + brow) * K + k0 + bkc;
        *(uint4*)&As[0][arow * LDA + akc]     = *(const uint4*)ag;
        *(uint4*)&As[0][arow * LDA + akc + 8] = *(const uint4*)(ag + 8);
        *(uint4*)&Bs[0][brow * LDA + bkc]     = *(const uint4*)bg;
        if (k0 + BK < K) {                    // CDNA5 global_prefetch path
            __builtin_prefetch(ag + BK, 0, 0);
            __builtin_prefetch(bg + BK, 0, 0);
        }
        __syncthreads();
        compute_tile(0);
        __syncthreads();
    }
#endif

    // epilogue: element (m,n) = (m0+wm+tm*16+r+8*half, n0+wn+tn*16+l16)
#pragma unroll
    for (int tm = 0; tm < 2; tm++)
#pragma unroll
        for (int tn = 0; tn < 2; tn++)
#pragma unroll
            for (int r = 0; r < 8; r++) {
                int m = m0 + wm + tm * 16 + r + half * 8;
                int n = n0 + wn + tn * 16 + l16;
                float val = acc[tm][tn][r] + bias[n];
                if constexpr (MODE == 0) {
                    out_bf[(size_t)m * Nout + n] = f2bf(val);
                } else if constexpr (MODE == 1) {
                    float g = 0.5f * val * (1.f + erff(val * 0.70710678118654752f));
                    out_bf[(size_t)m * Nout + n] = f2bf(g);
                } else if constexpr (MODE == 2) {
                    int win = m / N_, nidx = m % N_;
                    int bimg = win >> 6, wrem = win & 63;
                    int hh = (wrem >> 3) * WS_ + nidx / WS_;
                    int ww = (wrem & 7) * WS_ + nidx % WS_;
                    if (shift) { hh = (hh + WS_) % H_; ww = (ww + WS_) % H_; }
                    size_t idx = ((size_t)bimg * (H_ * H_) + hh * H_ + ww) * C_ + n;
                    xdst[idx] = xsrc[idx] + val;
                } else {
                    size_t idx = (size_t)m * C_ + n;
                    xdst[idx] = xsrc[idx] + val;
                }
            }
}

// ---------------------------------------------------------------------------
// Window attention: one block (4 waves) per (window, head).
// qkv layout: [MTOT][576] bf16, q=+0, k=+192, v=+384, head offset head*32.
// S = (Q K^T)*SCALE + relpos bias (+ shift mask); softmax; out = P V.
// N=49 padded to 64; pad logits forced to -1e30 so P pad cols are exact 0.
// ---------------------------------------------------------------------------
__global__ __launch_bounds__(128) void attn_kernel(
    const bf16_t* __restrict__ qkv, const float* __restrict__ rpb,
    bf16_t* __restrict__ out, int shift)
{
    int unit = blockIdx.x;
    int win = unit / NH_, head = unit % NH_;
    int wv = threadIdx.x >> 5, lane = threadIdx.x & 31;
    int half = lane >> 4, l16 = lane & 15;

    __shared__ bf16_t Ps[64 * 64];   // softmaxed probs, row-major, stride 64
    __shared__ bf16_t Vt[32 * 64];   // V transposed: [d][key], stride 64

    // stage V^T into LDS: thread t -> key t>>1, 16 d-values
    {
        int key = threadIdx.x >> 1;
        int dbase = (threadIdx.x & 1) * 16;
        int key_c = key < N_ ? key : (N_ - 1);
        const bf16_t* vrow = qkv + ((size_t)win * N_ + key_c) * 576 + 384 + head * HD_ + dbase;
        Frag tv;
        tv.q[0] = *(const uint4*)vrow;
        tv.q[1] = *(const uint4*)(vrow + 8);
        const bf16_t* te = (const bf16_t*)&tv;
#pragma unroll
        for (int j = 0; j < 16; j++) Vt[(dbase + j) * 64 + key] = te[j];
    }

    // ---- S = Q K^T : wave wv owns query rows [16wv, 16wv+16), 4 key tiles ----
    int qn = 16 * wv + l16;
    int qn_c = qn < N_ ? qn : (N_ - 1);
    const bf16_t* qrow = qkv + ((size_t)win * N_ + qn_c) * 576 + head * HD_;
    Frag aq;
    aq.q[0] = *(const uint4*)(qrow + half * 8);
    aq.q[1] = *(const uint4*)(qrow + 16 + half * 8);

    float logit[4][8];
    v8f zero = {0.f, 0.f, 0.f, 0.f, 0.f, 0.f, 0.f, 0.f};
#pragma unroll
    for (int tn = 0; tn < 4; tn++) {
        int kn = 16 * tn + l16;
        int kn_c = kn < N_ ? kn : (N_ - 1);
        const bf16_t* krow = qkv + ((size_t)win * N_ + kn_c) * 576 + 192 + head * HD_;
        Frag bk;
        bk.q[0] = *(const uint4*)(krow + half * 8);
        bk.q[1] = *(const uint4*)(krow + 16 + half * 8);
        v8f s = __builtin_amdgcn_wmma_f32_16x16x32_bf16(
            false, aq.v, false, bk.v, (short)0, zero, false, false);
#pragma unroll
        for (int r = 0; r < 8; r++) {
            int qm = 16 * wv + r + 8 * half;   // query index of this element
            float lv;
            if (kn >= N_ || qm >= N_) {
                lv = -1e30f;
            } else {
                int qy = qm / WS_, qx = qm % WS_;
                int ky = kn / WS_, kx = kn % WS_;
                float bia = rpb[((qy - ky + 6) * 13 + (qx - kx + 6)) * NH_ + head];
                float msk = 0.f;
                if (shift) {
                    int wrem = win & 63;
                    int wy = wrem >> 3, wx = wrem & 7;
                    int idq = regid(wy * WS_ + qy) * 3 + regid(wx * WS_ + qx);
                    int idk = regid(wy * WS_ + ky) * 3 + regid(wx * WS_ + kx);
                    if (idq != idk) msk = -100.f;
                }
                lv = s[r] * SCALE_ + bia + msk;
            }
            logit[tn][r] = lv;
        }
    }

    // ---- row softmax: row owned by 16 lanes (same half), 4 regs each ----
#pragma unroll
    for (int r = 0; r < 8; r++) {
        float mx = -1e30f;
#pragma unroll
        for (int tn = 0; tn < 4; tn++) mx = fmaxf(mx, logit[tn][r]);
#pragma unroll
        for (int o = 8; o >= 1; o >>= 1) mx = fmaxf(mx, __shfl_xor(mx, o, 32));
        float sum = 0.f;
#pragma unroll
        for (int tn = 0; tn < 4; tn++) {
            float e = __expf(logit[tn][r] - mx);
            logit[tn][r] = e;
            sum += e;
        }
#pragma unroll
        for (int o = 8; o >= 1; o >>= 1) sum += __shfl_xor(sum, o, 32);
        float inv = 1.f / sum;
        int qm = 16 * wv + r + 8 * half;
#pragma unroll
        for (int tn = 0; tn < 4; tn++)
            Ps[qm * 64 + 16 * tn + l16] = f2bf(logit[tn][r] * inv);
    }

    __syncthreads();

    // ---- out = P V : per wave 16x32 output, K=64 in two 32-steps ----
    v8f ao[2];
    ao[0] = zero; ao[1] = zero;
#pragma unroll
    for (int ks = 0; ks < 2; ks++) {
        Frag ap;
        const bf16_t* pp = &Ps[(16 * wv + l16) * 64 + 32 * ks + half * 8];
        ap.q[0] = *(const uint4*)pp;
        ap.q[1] = *(const uint4*)(pp + 16);
#pragma unroll
        for (int nd = 0; nd < 2; nd++) {
            Frag bv;
            const bf16_t* vp = &Vt[(16 * nd + l16) * 64 + 32 * ks + half * 8];
            bv.q[0] = *(const uint4*)vp;
            bv.q[1] = *(const uint4*)(vp + 16);
            ao[nd] = __builtin_amdgcn_wmma_f32_16x16x32_bf16(
                false, ap.v, false, bv.v, (short)0, ao[nd], false, false);
        }
    }
#pragma unroll
    for (int nd = 0; nd < 2; nd++)
#pragma unroll
        for (int r = 0; r < 8; r++) {
            int qm = 16 * wv + r + 8 * half;
            if (qm < N_) {
                int d = 16 * nd + l16;
                out[((size_t)win * N_ + qm) * C_ + head * HD_ + d] = f2bf(ao[nd][r]);
            }
        }
}

// ---------------------------------------------------------------------------
// Host-side orchestration
// ---------------------------------------------------------------------------
extern "C" void kernel_launch(void* const* d_in, const int* in_sizes, int n_in,
                              void* d_out, int out_size, void* d_ws, size_t ws_size,
                              hipStream_t stream) {
    const float* x     = (const float*)d_in[0];
    const float* n1w   = (const float*)d_in[1];
    const float* n1b   = (const float*)d_in[2];
    const float* qkvw  = (const float*)d_in[3];
    const float* qkvb  = (const float*)d_in[4];
    const float* rpb   = (const float*)d_in[5];
    const float* projw = (const float*)d_in[6];
    const float* projb = (const float*)d_in[7];
    const float* n2w   = (const float*)d_in[8];
    const float* n2b   = (const float*)d_in[9];
    const float* f1w   = (const float*)d_in[10];
    const float* f1b   = (const float*)d_in[11];
    const float* f2w   = (const float*)d_in[12];
    const float* f2b   = (const float*)d_in[13];

    const size_t xel = (size_t)MTOT * C_;          // 19,267,584
    char* ws = (char*)d_ws;
    float*  xbuf = (float*)ws;                                   // fp32 residual
    bf16_t* abuf = (bf16_t*)(ws + xel * 4);                      // bf16 [M][192]
    bf16_t* qbuf = (bf16_t*)(ws + xel * 4 + xel * 2);            // bf16 [M][768] max
    bf16_t* obuf = (bf16_t*)(ws + xel * 4 + xel * 2 + (size_t)MTOT * 768 * 2);
    bf16_t* wcvt = (bf16_t*)((char*)obuf + xel * 2);             // bf16 weights

    const int QKVW_SZ = 576 * 192, PROJW_SZ = 192 * 192;
    const int F1W_SZ = 768 * 192, F2W_SZ = 192 * 768;
    const int LSTRIDE = QKVW_SZ + PROJW_SZ + F1W_SZ + F2W_SZ;    // 442368

    (void)hipMemcpyAsync(xbuf, x, xel * 4, hipMemcpyDeviceToDevice, stream);

    for (int l = 0; l < 2; l++) {
        bf16_t* wl = wcvt + (size_t)l * LSTRIDE;
        cvt_kernel<<<(QKVW_SZ + 255) / 256, 256, 0, stream>>>(qkvw + (size_t)l * QKVW_SZ, wl, QKVW_SZ);
        cvt_kernel<<<(PROJW_SZ + 255) / 256, 256, 0, stream>>>(projw + (size_t)l * PROJW_SZ, wl + QKVW_SZ, PROJW_SZ);
        cvt_kernel<<<(F1W_SZ + 255) / 256, 256, 0, stream>>>(f1w + (size_t)l * F1W_SZ, wl + QKVW_SZ + PROJW_SZ, F1W_SZ);
        cvt_kernel<<<(F2W_SZ + 255) / 256, 256, 0, stream>>>(f2w + (size_t)l * F2W_SZ, wl + QKVW_SZ + PROJW_SZ + F1W_SZ, F2W_SZ);
    }

    const int MB = MTOT / BM;   // 784
    for (int layer = 0; layer < 2; layer++) {
        int shift = layer ? (WS_ / 2) : 0;
        const bf16_t* wl = wcvt + (size_t)layer * LSTRIDE;
        const bf16_t* wq = wl;
        const bf16_t* wp = wq + QKVW_SZ;
        const bf16_t* w1 = wp + PROJW_SZ;
        const bf16_t* w2 = w1 + F1W_SZ;

        // LN1 + shift + window partition
        ln_kernel<true><<<MTOT / 8, 256, 0, stream>>>(
            xbuf, n1w + layer * C_, n1b + layer * C_, abuf, shift);
        // QKV GEMM -> [M][576] bf16
        gemm_kernel<0><<<dim3(576 / BN, MB), 256, 0, stream>>>(
            abuf, wq, qkvb + layer * 576, MTOT, 192, 576,
            nullptr, nullptr, qbuf, 0);
        // window attention -> [M][192] bf16
        attn_kernel<<<BW_ * NH_, 128, 0, stream>>>(
            qbuf, rpb + (size_t)layer * 169 * NH_, obuf, shift);
        // proj GEMM + window-reverse + roll(+WS) + residual into xbuf
        gemm_kernel<2><<<dim3(192 / BN, MB), 256, 0, stream>>>(
            obuf, wp, projb + layer * C_, MTOT, 192, 192,
            xbuf, xbuf, nullptr, shift);
        // LN2 (token order)
        ln_kernel<false><<<MTOT / 8, 256, 0, stream>>>(
            xbuf, n2w + layer * C_, n2b + layer * C_, abuf, 0);
        // MLP fc1 + exact GELU -> [M][768] bf16
        gemm_kernel<1><<<dim3(768 / BN, MB), 256, 0, stream>>>(
            abuf, w1, f1b + layer * 768, MTOT, 192, 768,
            nullptr, nullptr, qbuf, 0);
        // MLP fc2 + residual; final layer writes d_out
        float* dst = layer ? (float*)d_out : xbuf;
        gemm_kernel<3><<<dim3(192 / BN, MB), 256, 0, stream>>>(
            qbuf, w2, f2b + layer * C_, MTOT, 768, 192,
            xbuf, dst, nullptr, 0);
    }
}
